// CantorRelationalBlock_84052509983127
// MI455X (gfx1250) — matrix operand order repaired and checked by reading, not compiled, not run on hardware
//
#include <hip/hip_runtime.h>
#include <math.h>

// ---------------------------------------------------------------------------
// CantorRelationalBlock on MI455X (gfx1250).
// All dense matmuls: v_wmma_f32_16x16x32_bf16 (bf16 in, f32 acc), fed by
// global_load_async_to_lds_b128 double-buffered staging (ASYNCcnt pipeline).
// Weights are transpose-converted to bf16 [N x K] once per launch; every
// GEMM input is produced directly in bf16 by its producer kernel.
// ---------------------------------------------------------------------------

#define BATCH    2
#define SEQ      1024
#define DIMS     768
#define HEADS    12
#define HEAD_DIM 64
#define WIN      64
#define MROWS    (BATCH * SEQ)   // 2048

#define TILE_M 128
#define TILE_N 64
#define KSTEP  32

typedef __attribute__((ext_vector_type(16))) __bf16       v16bf;
typedef __attribute__((ext_vector_type(8)))  float        v8f;
typedef __attribute__((ext_vector_type(4)))  unsigned int u32x4;
typedef __attribute__((ext_vector_type(4)))  float        f32x4;

union BF16Frag {
  v16bf v;
  u32x4 q[2];
  unsigned short u[16];
};

__device__ __forceinline__ unsigned short f2bf(float f) {
  unsigned int u = __float_as_uint(f);
  u += 0x7FFFu + ((u >> 16) & 1u);   // round-to-nearest-even
  return (unsigned short)(u >> 16);
}

__device__ __forceinline__ float gelu_exact(float x) {
  return 0.5f * x * (1.0f + erff(x * 0.7071067811865475f));
}

// Async global -> LDS 16-byte copy (gfx1250). lds_off = wave-relative LDS byte
// address (low 32 bits of the generic LDS pointer).
__device__ __forceinline__ void async_cp16(unsigned lds_off, const void* gsrc) {
  asm volatile("global_load_async_to_lds_b128 %0, %1, off"
               :: "v"(lds_off), "v"(gsrc)
               : "memory");
}

// ---------------------------------------------------------------------------
// Weight transpose + convert: W [K x N] f32  ->  WT [N x K] bf16.
// 64x64 tiles through LDS; both global sides coalesced. K,N multiples of 64.
// ---------------------------------------------------------------------------
__global__ __launch_bounds__(256) void transpose_convert_kernel(
    const float* __restrict__ W, unsigned short* __restrict__ WT, int K, int N) {
  __shared__ unsigned short t[64][64 + 4];
  const int tid = threadIdx.x;
  const int k0 = blockIdx.y * 64, n0 = blockIdx.x * 64;
  const int kk = tid >> 4;            // 0..15
  const int nn = (tid & 15) * 4;      // 0..60
#pragma unroll
  for (int c = 0; c < 4; ++c) {
    const int k = kk + c * 16;
    const f32x4 v = *(const f32x4*)(W + (size_t)(k0 + k) * N + n0 + nn);
    t[nn + 0][k] = f2bf(v.x);
    t[nn + 1][k] = f2bf(v.y);
    t[nn + 2][k] = f2bf(v.z);
    t[nn + 3][k] = f2bf(v.w);
  }
  __syncthreads();
  const int k2 = (tid & 15) * 4;
  const int n2 = tid >> 4;
#pragma unroll
  for (int c = 0; c < 4; ++c) {
    const int n = n2 + c * 16;
    unsigned short* dst = WT + (size_t)(n0 + n) * K + k0 + k2;
    dst[0] = t[n][k2 + 0];
    dst[1] = t[n][k2 + 1];
    dst[2] = t[n][k2 + 2];
    dst[3] = t[n][k2 + 3];
  }
}

// ---------------------------------------------------------------------------
// GEMM: Y[M,N] = act(Xbf16[M,K] @ WT16[N,K]^T + bias) (+ res).
// 256 threads (8 waves), tile 128x64, K step 32. Double-buffered LDS filled by
// global_load_async_to_lds_b128 (3 x 16B per thread per tile). Fragments read
// with ds_load_b128 in contiguous k order (k = 16*half + s) for A and B; 4 WMMAs back-to-back.
// Optional fp32 and/or bf16 stores.
// ---------------------------------------------------------------------------
__global__ __launch_bounds__(256) void gemm_bf16_wmma(
    const unsigned short* __restrict__ X,   // bf16 [M x K]
    const unsigned short* __restrict__ WT,  // bf16 [N x K]  (W transposed)
    const float* __restrict__ bias,
    const float* __restrict__ res,
    float* __restrict__ Yf, unsigned short* __restrict__ Y16,
    int M, int N, int K, int act /*1=gelu*/) {
  __shared__ unsigned short sA[2][TILE_M * KSTEP]; // [m][k] bf16, ping-pong
  __shared__ unsigned short sB[2][TILE_N * KSTEP]; // [n][k] bf16, ping-pong

  const int tid  = threadIdx.x;
  const int wave = tid >> 5;
  const int lane = tid & 31;
  const int m0 = blockIdx.y * TILE_M;
  const int n0 = blockIdx.x * TILE_N;
  const int l16  = lane & 15;
  const int half = lane >> 4;

  v8f acc[4];
#pragma unroll
  for (int j = 0; j < 4; ++j) {
    const float bv = bias[n0 + j * 16 + l16];
#pragma unroll
    for (int i = 0; i < 8; ++i) acc[j][i] = bv;
  }

  // per-thread async copy coordinates (3 x 16B chunks per tile)
  const int arow0 = tid >> 2,          aco0 = (tid & 3) * 8;         // A chunk 0
  const int arow1 = (tid + 256) >> 2,  aco1 = ((tid + 256) & 3) * 8; // A chunk 1
  const int brow  = tid >> 2,          bco  = (tid & 3) * 8;         // B chunk

  auto fill = [&](int bf, int k0) {
    const unsigned aBase = (unsigned)(unsigned long long)(&sA[bf][0]);
    const unsigned bBase = (unsigned)(unsigned long long)(&sB[bf][0]);
    async_cp16(aBase + (unsigned)(arow0 * KSTEP + aco0) * 2,
               X + (size_t)(m0 + arow0) * K + k0 + aco0);
    async_cp16(aBase + (unsigned)(arow1 * KSTEP + aco1) * 2,
               X + (size_t)(m0 + arow1) * K + k0 + aco1);
    async_cp16(bBase + (unsigned)(brow * KSTEP + bco) * 2,
               WT + (size_t)(n0 + brow) * K + k0 + bco);
  };

  const int KT = K / KSTEP;
  fill(0, 0);
  for (int kt = 0; kt < KT; ++kt) {
    const int cur = kt & 1;
    if (kt + 1 < KT) {
      fill(cur ^ 1, (kt + 1) * KSTEP);
      asm volatile("s_wait_asynccnt 3" ::: "memory"); // current tile's 3 done
    } else {
      asm volatile("s_wait_asynccnt 0" ::: "memory");
    }
    __syncthreads();

    const unsigned short* pA = &sA[cur][0];
    const unsigned short* pB = &sB[cur][0];
    // A 16x32 bf16: lane m=l16; K = half*16 + 0..15 contiguous in [m][k].
    BF16Frag a, b0, b1, b2, b3;
    const int mr = wave * 16 + l16;
    const int kh = half * 16;
    a.q[0] = *(const u32x4*)&pA[mr * KSTEP + kh];
    a.q[1] = *(const u32x4*)&pA[mr * KSTEP + kh + 8];
    // B 32x16 bf16: lane n=l16; K = half*16 + 0..15 contiguous in [n][k].
    const int kb = half * 16;
    b0.q[0] = *(const u32x4*)&pB[(0 * 16 + l16) * KSTEP + kb];
    b0.q[1] = *(const u32x4*)&pB[(0 * 16 + l16) * KSTEP + kb + 8];
    b1.q[0] = *(const u32x4*)&pB[(1 * 16 + l16) * KSTEP + kb];
    b1.q[1] = *(const u32x4*)&pB[(1 * 16 + l16) * KSTEP + kb + 8];
    b2.q[0] = *(const u32x4*)&pB[(2 * 16 + l16) * KSTEP + kb];
    b2.q[1] = *(const u32x4*)&pB[(2 * 16 + l16) * KSTEP + kb + 8];
    b3.q[0] = *(const u32x4*)&pB[(3 * 16 + l16) * KSTEP + kb];
    b3.q[1] = *(const u32x4*)&pB[(3 * 16 + l16) * KSTEP + kb + 8];

    acc[0] = __builtin_amdgcn_wmma_f32_16x16x32_bf16(false, a.v, false, b0.v,
                                                     (short)0, acc[0], false, false);
    acc[1] = __builtin_amdgcn_wmma_f32_16x16x32_bf16(false, a.v, false, b1.v,
                                                     (short)0, acc[1], false, false);
    acc[2] = __builtin_amdgcn_wmma_f32_16x16x32_bf16(false, a.v, false, b2.v,
                                                     (short)0, acc[2], false, false);
    acc[3] = __builtin_amdgcn_wmma_f32_16x16x32_bf16(false, a.v, false, b3.v,
                                                     (short)0, acc[3], false, false);
    __syncthreads();
  }

  // C/D layout: VGPR i -> row half*8+i, col l16.
  const int mbase = m0 + wave * 16 + half * 8;
#pragma unroll
  for (int j = 0; j < 4; ++j) {
    const int col = n0 + j * 16 + l16;
#pragma unroll
    for (int i = 0; i < 8; ++i) {
      const size_t idx = (size_t)(mbase + i) * N + col;
      float v = acc[j][i];
      if (act == 1) v = gelu_exact(v);
      if (res) v += res[idx];
      if (Yf) Yf[idx] = v;
      if (Y16) Y16[idx] = f2bf(v);
    }
  }
}

// ---------------------------------------------------------------------------
// LayerNorm over last dim (768), fp32 in -> bf16 out (feeds GEMMs only).
// ---------------------------------------------------------------------------
__global__ __launch_bounds__(256) void layernorm_kernel(
    const float* __restrict__ X, const float* __restrict__ g,
    const float* __restrict__ b, unsigned short* __restrict__ Y16) {
  const int row = blockIdx.x;
  const int tid = threadIdx.x;
  const float* x = X + (size_t)row * DIMS;
  unsigned short* y = Y16 + (size_t)row * DIMS;
  __shared__ float red[256];

  const float v0 = x[tid], v1 = x[tid + 256], v2 = x[tid + 512];
  red[tid] = v0 + v1 + v2;
  __syncthreads();
  for (int st = 128; st > 0; st >>= 1) {
    if (tid < st) red[tid] += red[tid + st];
    __syncthreads();
  }
  const float mean = red[0] * (1.0f / DIMS);
  __syncthreads();
  const float d0 = v0 - mean, d1 = v1 - mean, d2 = v2 - mean;
  red[tid] = d0 * d0 + d1 * d1 + d2 * d2;
  __syncthreads();
  for (int st = 128; st > 0; st >>= 1) {
    if (tid < st) red[tid] += red[tid + st];
    __syncthreads();
  }
  const float rstd = rsqrtf(red[0] * (1.0f / DIMS) + 1e-5f);
  y[tid]       = f2bf(d0 * rstd * g[tid]       + b[tid]);
  y[tid + 256] = f2bf(d1 * rstd * g[tid + 256] + b[tid + 256]);
  y[tid + 512] = f2bf(d2 * rstd * g[tid + 512] + b[tid + 512]);
}

// ---------------------------------------------------------------------------
// Sparse neighbor attention (fp32 math). Block: (b, h, 4 queries).
// Output written in bf16, head-merged [row, h*64+d] (feeds out-proj GEMM).
// ---------------------------------------------------------------------------
__global__ __launch_bounds__(256) void sparse_attn_kernel(
    const float* __restrict__ Qb, int ldq, int qoff,
    const float* __restrict__ Kb, int ldk, int koff,
    const float* __restrict__ Vb, int ldv, int voff,
    const int* __restrict__ neighbors, unsigned short* __restrict__ Out16) {
  const int h  = blockIdx.y;
  const int bb = blockIdx.z;
  const int tid = threadIdx.x;
  const int qi = tid >> 6;
  const int w  = tid & 63;
  const int s  = blockIdx.x * 4 + qi;
  const int qrow = bb * SEQ + s;

  __shared__ float qs[4][WIN];
  __shared__ int   nbs[4][WIN];
  __shared__ float sc[4][WIN];
  __shared__ float pe[4][WIN];

  nbs[qi][w] = neighbors[s * WIN + w];
  qs[qi][w]  = Qb[(size_t)qrow * ldq + qoff + h * HEAD_DIM + w] * 0.125f;
  __syncthreads();

  {
    const int n = nbs[qi][w];
    const float* kp = Kb + (size_t)(bb * SEQ + n) * ldk + koff + h * HEAD_DIM;
    float acc = 0.0f;
#pragma unroll 8
    for (int d = 0; d < HEAD_DIM; ++d) acc += qs[qi][d] * kp[d];
    sc[qi][w] = acc;
  }
  __syncthreads();

  float m = -3.0e38f;
#pragma unroll 8
  for (int j = 0; j < WIN; ++j) m = fmaxf(m, sc[qi][j]);
  const float p = expf(sc[qi][w] - m);
  pe[qi][w] = p;
  __syncthreads();
  float denom = 0.0f;
#pragma unroll 8
  for (int j = 0; j < WIN; ++j) denom += pe[qi][j];
  sc[qi][w] = p / denom;
  __syncthreads();

  {
    const int d = w;
    float o = 0.0f;
    for (int j = 0; j < WIN; ++j) {
      const int n = nbs[qi][j];
      o += sc[qi][j] * Vb[(size_t)(bb * SEQ + n) * ldv + voff + h * HEAD_DIM + d];
    }
    Out16[(size_t)qrow * DIMS + h * HEAD_DIM + d] = f2bf(o);
  }
}

// ---------------------------------------------------------------------------
// Host orchestration
// ---------------------------------------------------------------------------
extern "C" void kernel_launch(void* const* d_in, const int* in_sizes, int n_in,
                              void* d_out, int out_size, void* d_ws, size_t ws_size,
                              hipStream_t stream) {
  (void)in_sizes; (void)n_in; (void)out_size; (void)ws_size;

  // setup_inputs() dict insertion order.
  const float* clip_embed = (const float*)d_in[0];
  const float* t5_embed   = (const float*)d_in[1];
  const float* clip_sa_qkv_w = (const float*)d_in[2];
  const float* clip_sa_qkv_b = (const float*)d_in[3];
  const float* clip_sa_out_w = (const float*)d_in[4];
  const float* clip_sa_out_b = (const float*)d_in[5];
  const float* t5_sa_qkv_w   = (const float*)d_in[6];
  const float* t5_sa_qkv_b   = (const float*)d_in[7];
  const float* t5_sa_out_w   = (const float*)d_in[8];
  const float* t5_sa_out_b   = (const float*)d_in[9];
  const float* t5_ca_q_w   = (const float*)d_in[10];
  const float* t5_ca_q_b   = (const float*)d_in[11];
  const float* t5_ca_kv_w  = (const float*)d_in[12];
  const float* t5_ca_kv_b  = (const float*)d_in[13];
  const float* t5_ca_out_w = (const float*)d_in[14];
  const float* t5_ca_out_b = (const float*)d_in[15];
  const float* clip_ca_q_w   = (const float*)d_in[16];
  const float* clip_ca_q_b   = (const float*)d_in[17];
  const float* clip_ca_kv_w  = (const float*)d_in[18];
  const float* clip_ca_kv_b  = (const float*)d_in[19];
  const float* clip_ca_out_w = (const float*)d_in[20];
  const float* clip_ca_out_b = (const float*)d_in[21];
  const float* clip_ffn_w1 = (const float*)d_in[22];
  const float* clip_ffn_b1 = (const float*)d_in[23];
  const float* clip_ffn_w2 = (const float*)d_in[24];
  const float* clip_ffn_b2 = (const float*)d_in[25];
  const float* t5_ffn_w1 = (const float*)d_in[26];
  const float* t5_ffn_b1 = (const float*)d_in[27];
  const float* t5_ffn_w2 = (const float*)d_in[28];
  const float* t5_ffn_b2 = (const float*)d_in[29];
  const float* clip_n1_g = (const float*)d_in[30];
  const float* clip_n1_b = (const float*)d_in[31];
  const float* clip_n2_g = (const float*)d_in[32];
  const float* clip_n2_b = (const float*)d_in[33];
  const float* clip_n3_g = (const float*)d_in[34];
  const float* clip_n3_b = (const float*)d_in[35];
  const float* t5_n1_g = (const float*)d_in[36];
  const float* t5_n1_b = (const float*)d_in[37];
  const float* t5_n2_g = (const float*)d_in[38];
  const float* t5_n2_b = (const float*)d_in[39];
  const float* t5_n3_g = (const float*)d_in[40];
  const float* t5_n3_b = (const float*)d_in[41];
  const int* neighbors = (const int*)d_in[42];

  float* out_clip = (float*)d_out;
  float* out_t5   = (float*)d_out + (size_t)MROWS * DIMS;

  // ---- workspace (bump allocator, 256B aligned chunks; ~84 MiB total) ----
  char* p = (char*)d_ws;
  auto alloc = [&](size_t bytes) {
    char* r = p;
    p += (bytes + 255) & ~(size_t)255;
    return r;
  };
  typedef unsigned short us;
  us* wt_qkv_c  = (us*)alloc((size_t)768 * 2304 * 2);
  us* wt_out_c  = (us*)alloc((size_t)768 * 768 * 2);
  us* wt_qkv_t  = (us*)alloc((size_t)768 * 2304 * 2);
  us* wt_out_t  = (us*)alloc((size_t)768 * 768 * 2);
  us* wt_q_t5   = (us*)alloc((size_t)768 * 768 * 2);
  us* wt_kv_t5  = (us*)alloc((size_t)768 * 1536 * 2);
  us* wt_out_t5 = (us*)alloc((size_t)768 * 768 * 2);
  us* wt_q_c    = (us*)alloc((size_t)768 * 768 * 2);
  us* wt_kv_c   = (us*)alloc((size_t)768 * 1536 * 2);
  us* wt_out_ca = (us*)alloc((size_t)768 * 768 * 2);
  us* wt_w1_c   = (us*)alloc((size_t)768 * 3072 * 2);
  us* wt_w2_c   = (us*)alloc((size_t)3072 * 768 * 2);
  us* wt_w1_t   = (us*)alloc((size_t)768 * 3072 * 2);
  us* wt_w2_t   = (us*)alloc((size_t)3072 * 768 * 2);
  us*    ln16        = (us*)alloc((size_t)MROWS * DIMS * 2);
  us*    attn16      = (us*)alloc((size_t)MROWS * DIMS * 2);
  float* qbuf        = (float*)alloc((size_t)MROWS * DIMS * 4);
  float* bigf        = (float*)alloc((size_t)MROWS * 3 * DIMS * 4);
  float* clip_self   = (float*)alloc((size_t)MROWS * DIMS * 4);
  float* t5_self     = (float*)alloc((size_t)MROWS * DIMS * 4);
  us*    clip_self16 = (us*)alloc((size_t)MROWS * DIMS * 2);
  us*    t5_self16   = (us*)alloc((size_t)MROWS * DIMS * 2);
  us*    hidden16    = (us*)bigf;   // alias: ffn stage no longer needs bigf
  float* clip_cross  = out_clip;    // residual read-then-overwrite, same thread
  float* t5_cross    = out_t5;

  auto tc = [&](const float* W, us* WT, int K, int N) {
    transpose_convert_kernel<<<dim3(N / 64, K / 64), 256, 0, stream>>>(W, WT, K, N);
  };
  auto ln = [&](const float* x, const float* g, const float* b, us* y) {
    layernorm_kernel<<<dim3(MROWS), 256, 0, stream>>>(x, g, b, y);
  };
  auto gemm = [&](const us* X, const us* WT, const float* bias, const float* res,
                  float* Yf, us* Y16, int N, int K, int act) {
    gemm_bf16_wmma<<<dim3(N / TILE_N, MROWS / TILE_M), 256, 0, stream>>>(
        X, WT, bias, res, Yf, Y16, MROWS, N, K, act);
  };
  auto attn = [&](const float* Q, int ldq, int qo, const float* K, int ldk, int ko,
                  const float* V, int ldv, int vo, us* O) {
    sparse_attn_kernel<<<dim3(SEQ / 4, HEADS, BATCH), 256, 0, stream>>>(
        Q, ldq, qo, K, ldk, ko, V, ldv, vo, neighbors, O);
  };

  // ---- one-time (per launch) weight transpose+convert to bf16 [N x K] ----
  tc(clip_sa_qkv_w, wt_qkv_c, DIMS, 3 * DIMS);
  tc(clip_sa_out_w, wt_out_c, DIMS, DIMS);
  tc(t5_sa_qkv_w,   wt_qkv_t, DIMS, 3 * DIMS);
  tc(t5_sa_out_w,   wt_out_t, DIMS, DIMS);
  tc(t5_ca_q_w,     wt_q_t5,  DIMS, DIMS);
  tc(t5_ca_kv_w,    wt_kv_t5, DIMS, 2 * DIMS);
  tc(t5_ca_out_w,   wt_out_t5, DIMS, DIMS);
  tc(clip_ca_q_w,   wt_q_c,   DIMS, DIMS);
  tc(clip_ca_kv_w,  wt_kv_c,  DIMS, 2 * DIMS);
  tc(clip_ca_out_w, wt_out_ca, DIMS, DIMS);
  tc(clip_ffn_w1,   wt_w1_c,  DIMS, 4 * DIMS);
  tc(clip_ffn_w2,   wt_w2_c,  4 * DIMS, DIMS);
  tc(t5_ffn_w1,     wt_w1_t,  DIMS, 4 * DIMS);
  tc(t5_ffn_w2,     wt_w2_t,  4 * DIMS, DIMS);

  // ---- CLIP self-attention ----
  ln(clip_embed, clip_n1_g, clip_n1_b, ln16);
  gemm(ln16, wt_qkv_c, clip_sa_qkv_b, nullptr, bigf, nullptr, 3 * DIMS, DIMS, 0);
  attn(bigf, 3 * DIMS, 0, bigf, 3 * DIMS, DIMS, bigf, 3 * DIMS, 2 * DIMS, attn16);
  gemm(attn16, wt_out_c, clip_sa_out_b, clip_embed, clip_self, clip_self16, DIMS, DIMS, 0);

  // ---- T5 self-attention ----
  ln(t5_embed, t5_n1_g, t5_n1_b, ln16);
  gemm(ln16, wt_qkv_t, t5_sa_qkv_b, nullptr, bigf, nullptr, 3 * DIMS, DIMS, 0);
  attn(bigf, 3 * DIMS, 0, bigf, 3 * DIMS, DIMS, bigf, 3 * DIMS, 2 * DIMS, attn16);
  gemm(attn16, wt_out_t, t5_sa_out_b, t5_embed, t5_self, t5_self16, DIMS, DIMS, 0);

  // ---- T5 cross-attention (q from LN(t5_self), kv from raw clip_self) ----
  ln(t5_self, t5_n2_g, t5_n2_b, ln16);
  gemm(ln16, wt_q_t5, t5_ca_q_b, nullptr, qbuf, nullptr, DIMS, DIMS, 0);
  gemm(clip_self16, wt_kv_t5, t5_ca_kv_b, nullptr, bigf, nullptr, 2 * DIMS, DIMS, 0);
  attn(qbuf, DIMS, 0, bigf, 2 * DIMS, 0, bigf, 2 * DIMS, DIMS, attn16);
  gemm(attn16, wt_out_t5, t5_ca_out_b, t5_self, t5_cross, nullptr, DIMS, DIMS, 0);

  // ---- CLIP cross-attention (q from LN(clip_self), kv from raw t5_self) ----
  ln(clip_self, clip_n2_g, clip_n2_b, ln16);
  gemm(ln16, wt_q_c, clip_ca_q_b, nullptr, qbuf, nullptr, DIMS, DIMS, 0);
  gemm(t5_self16, wt_kv_c, clip_ca_kv_b, nullptr, bigf, nullptr, 2 * DIMS, DIMS, 0);
  attn(qbuf, DIMS, 0, bigf, 2 * DIMS, 0, bigf, 2 * DIMS, DIMS, attn16);
  gemm(attn16, wt_out_ca, clip_ca_out_b, clip_self, clip_cross, nullptr, DIMS, DIMS, 0);

  // ---- CLIP FFN ----
  ln(clip_cross, clip_n3_g, clip_n3_b, ln16);
  gemm(ln16, wt_w1_c, clip_ffn_b1, nullptr, nullptr, hidden16, 4 * DIMS, DIMS, 1);
  gemm(hidden16, wt_w2_c, clip_ffn_b2, clip_cross, out_clip, nullptr, DIMS, 4 * DIMS, 0);

  // ---- T5 FFN ----
  ln(t5_cross, t5_n3_g, t5_n3_b, ln16);
  gemm(ln16, wt_w1_t, t5_ffn_b1, nullptr, nullptr, hidden16, 4 * DIMS, DIMS, 1);
  gemm(hidden16, wt_w2_t, t5_ffn_b2, t5_cross, out_t5, nullptr, DIMS, 4 * DIMS, 0);
}
